// PCILTConv2D_25409026523664
// MI455X (gfx1250) — compile-verified, matrix operand order checked by reading, not run on hardware
//
#include <hip/hip_runtime.h>

typedef __attribute__((ext_vector_type(8))) int v8i;
typedef __attribute__((ext_vector_type(2))) int v2i;

#define N_IMG   16
#define C_IN    64
#define C_OUT   128
#define HW      112
#define WTILES  7          // 112 / 16
#define ROWS    4          // output rows per block
#define HTILES  (HW / ROWS)
#define QMAX    127.0f

// ---------------------------------------------------------------------------
// ws layout:
//   [0..4)   : uint bits of max|x|
//   [4..8)   : uint bits of max|W|
//   [64 ..)  : xq  int8 NHWC  : 16*112*112*64 = 12,845,056 B
//   [+ ..)   : wpack int8     : 8 octiles * 9 taps * 32 lanes * 32 B = 73,728 B
// ---------------------------------------------------------------------------
#define XQ_OFF   64
#define XQ_BYTES (N_IMG * HW * HW * C_IN)
#define WP_OFF   (XQ_OFF + XQ_BYTES)

__global__ void init_scales_kernel(unsigned* s) {
    if (threadIdx.x == 0) { s[0] = 0u; s[1] = 0u; }
}

// Grid-stride abs-max; |v| >= 0 so IEEE bits compare like unsigned ints.
__global__ void absmax_kernel(const float* __restrict__ p, long n, unsigned* __restrict__ dst) {
    __shared__ float red[256];
    float m = 0.0f;
    long stride = (long)gridDim.x * blockDim.x;
    for (long i = (long)blockIdx.x * blockDim.x + threadIdx.x; i < n; i += stride)
        m = fmaxf(m, fabsf(p[i]));
    red[threadIdx.x] = m;
    __syncthreads();
    for (int o = 128; o > 0; o >>= 1) {
        if ((int)threadIdx.x < o) red[threadIdx.x] = fmaxf(red[threadIdx.x], red[threadIdx.x + o]);
        __syncthreads();
    }
    if (threadIdx.x == 0) atomicMax(dst, __float_as_uint(red[0]));
}

// Quantize x (NCHW f32) -> xq (NHWC int8). One block per (n, h, 16-w tile).
// Reads strided over c, transposes through LDS, writes coalesced dwords.
__global__ void quant_x_kernel(const float* __restrict__ x, const unsigned* __restrict__ sc,
                               signed char* __restrict__ xq) {
    __shared__ __align__(16) signed char lds[16 * C_IN];  // [w][c]
    int bid = blockIdx.x;
    int wt = bid % WTILES;  int t = bid / WTILES;
    int h  = t % HW;        int n = t / HW;
    int w0 = wt * 16;
    float sx  = fmaxf(__uint_as_float(sc[0]) / QMAX, 1e-8f);
    float inv = 1.0f / sx;
    int tid = threadIdx.x;
#pragma unroll
    for (int i = 0; i < 4; ++i) {
        int e = tid + i * 256;          // 0..1023
        int c = e >> 4, w = e & 15;
        float v = x[(((long)n * C_IN + c) * HW + h) * HW + w0 + w];
        int q = (int)rintf(v * inv);
        q = q < -127 ? -127 : (q > 127 ? 127 : q);
        lds[w * C_IN + c] = (signed char)q;
    }
    __syncthreads();
    ((int*)xq)[((long)(n * HW + h) * HW + w0) * (C_IN / 4) + tid] = ((const int*)lds)[tid];
}

// Quantize + pack W (OIHW f32) -> per-lane A fragments matching the CDNA5
// 8-bit 16x64 A layout: M = lane&15; K chunks {0,16,32,48}+8*(lane>=16), 8B each.
__global__ void pack_w_kernel(const float* __restrict__ W, const unsigned* __restrict__ sc,
                              signed char* __restrict__ wpack) {
    int idx = blockIdx.x * 256 + threadIdx.x;
    if (idx >= 8 * 9 * 32 * 32) return;
    int byte   = idx & 31;
    int lane   = (idx >> 5) & 31;
    int tapid  = (idx >> 10) % 9;
    int octile = idx / (9 * 1024);
    int chunk = byte >> 3, within = byte & 7;
    int K = chunk * 16 + ((lane >> 4) << 3) + within;   // input channel
    int M = octile * 16 + (lane & 15);                  // output channel
    int r = tapid / 3, s = tapid % 3;
    float sw = fmaxf(__uint_as_float(sc[1]) / QMAX, 1e-8f);
    float v = W[((M * C_IN + K) * 3 + r) * 3 + s];
    int q = (int)rintf(v / sw);
    q = q < -127 ? -127 : (q > 127 ? 127 : q);
    wpack[idx] = (signed char)q;
}

// Implicit-GEMM int8 conv. One block (8 waves) per (n, 4-h rows, 16-w tile);
// wave w owns output channels [16w, 16w+16). A fragments are loaded once and
// reused across 4 output rows -> 36 x v_wmma_i32_16x16x64_iu8 per wave.
__global__ void __launch_bounds__(256)
conv_wmma_kernel(const signed char* __restrict__ xq, const signed char* __restrict__ wpack,
                 const unsigned* __restrict__ sc, float* __restrict__ out) {
    __shared__ __align__(16) signed char tile[(ROWS + 2) * 18 * C_IN];  // 6912 B halo tile
    int bid = blockIdx.x;
    int wt = bid % WTILES;  int t = bid / WTILES;
    int ht = t % HTILES;    int n = t / HTILES;
    int h0 = ht * ROWS;
    int w0 = wt * 16;
    int tid = threadIdx.x;

    // Stage halo tile (rows h0-1..h0+ROWS, cols w0-1..w0+16, all 64 c), zero pad.
    const int TILE_DW = (ROWS + 2) * 18 * (C_IN / 4);  // 1728
    for (int d = tid; d < TILE_DW; d += 256) {
        int pix = d >> 4, c4 = d & 15;
        int row = pix / 18, col = pix % 18;
        int hh = h0 + row - 1, ww = w0 + col - 1;
        int val = 0;
        if (hh >= 0 && hh < HW && ww >= 0 && ww < HW)
            val = ((const int*)xq)[(((long)(n * HW + hh) * HW + ww) << 4) + c4];
        ((int*)tile)[d] = val;
    }
    __syncthreads();

    int wave = tid >> 5;        // oc tile 0..7
    int lane = tid & 31;
    int half = lane >> 4, l15 = lane & 15;

    // Load all 9 A fragments once; reused for every output row.
    v8i a[9];
#pragma unroll
    for (int tapid = 0; tapid < 9; ++tapid)
        a[tapid] = *(const v8i*)(wpack + (((wave * 9 + tapid) * 32 + lane) << 5));

    float sx = fmaxf(__uint_as_float(sc[0]) / QMAX, 1e-8f);
    float sw = fmaxf(__uint_as_float(sc[1]) / QMAX, 1e-8f);
    float scale = sx * sw;

#pragma unroll
    for (int row = 0; row < ROWS; ++row) {
        v8i acc = (v8i)0;
#pragma unroll
        for (int r = 0; r < 3; ++r) {
#pragma unroll
            for (int s = 0; s < 3; ++s) {
                // B fragment from LDS, mirroring the 8-bit K layout.
                int pixoff = ((row + r) * 18 + (l15 + s)) * C_IN + half * 8;
                v2i c0 = *(const v2i*)(tile + pixoff + 0);
                v2i c1 = *(const v2i*)(tile + pixoff + 16);
                v2i c2 = *(const v2i*)(tile + pixoff + 32);
                v2i c3 = *(const v2i*)(tile + pixoff + 48);
                v8i b = {c0.x, c0.y, c1.x, c1.y, c2.x, c2.y, c3.x, c3.y};
                // (sgn_a, A, sgn_b, B, C, reuse_a, reuse_b)
                acc = __builtin_amdgcn_wmma_i32_16x16x64_iu8(true, a[r * 3 + s], true, b,
                                                             acc, false, false);
            }
        }
        // D layout: lane -> N=l15 (width), VGPR g -> M = g + 8*half (oc in tile).
        int oc0 = wave * 16 + 8 * half;
        long obase = (((long)n * C_OUT + oc0) * HW + (h0 + row)) * HW + w0 + l15;
#pragma unroll
        for (int g = 0; g < 8; ++g)
            out[obase + (long)g * HW * HW] = (float)acc[g] * scale;
    }
}

extern "C" void kernel_launch(void* const* d_in, const int* in_sizes, int n_in,
                              void* d_out, int out_size, void* d_ws, size_t ws_size,
                              hipStream_t stream) {
    const float* x = (const float*)d_in[0];   // [16,64,112,112]
    const float* W = (const float*)d_in[1];   // [128,64,3,3]
    float* out = (float*)d_out;               // [16,128,112,112]
    char* ws = (char*)d_ws;

    unsigned*    scales = (unsigned*)ws;
    signed char* xq     = (signed char*)(ws + XQ_OFF);
    signed char* wpack  = (signed char*)(ws + WP_OFF);

    const long nx = (long)N_IMG * C_IN * HW * HW;
    const long nw = (long)C_OUT * C_IN * 9;

    init_scales_kernel<<<1, 32, 0, stream>>>(scales);
    absmax_kernel<<<2048, 256, 0, stream>>>(x, nx, scales + 0);
    absmax_kernel<<<64,   256, 0, stream>>>(W, nw, scales + 1);
    quant_x_kernel<<<N_IMG * HW * WTILES, 256, 0, stream>>>(x, scales, xq);
    pack_w_kernel<<<(8 * 9 * 32 * 32 + 255) / 256, 256, 0, stream>>>(W, scales, wpack);
    conv_wmma_kernel<<<N_IMG * HTILES * WTILES, 256, 0, stream>>>(xq, wpack, scales, out);
}
